// GAT_90778428768714
// MI455X (gfx1250) — matrix operand order repaired and checked by reading, not compiled, not run on hardware
//
#include <hip/hip_runtime.h>

typedef __attribute__((ext_vector_type(16))) _Float16 v16h;
typedef __attribute__((ext_vector_type(8)))  _Float16 v8h;
typedef __attribute__((ext_vector_type(8)))  float    v8f;

#define F_IN 256
#define HEADS1 8
#define CH1 8
#define D1 64   /* HEADS1*CH1 */
#define D2 64
#define NEG_SLOPE 0.2f

// ---- monotone float<->uint mapping for atomic max on floats ----
static __device__ __forceinline__ unsigned f2o(float f) {
  unsigned u = __float_as_uint(f);
  return (u >> 31) ? ~u : (u ^ 0x80000000u);
}
static __device__ __forceinline__ float o2f(unsigned o) {
  return __uint_as_float((o >> 31) ? (o ^ 0x80000000u) : ~o);
}

// ---------------- elementwise f32 -> f16 ----------------
__global__ void cvt_f16_kernel(const float* __restrict__ in, _Float16* __restrict__ out, long n) {
  long i = (long)blockIdx.x * blockDim.x + threadIdx.x;
  if (i < n) out[i] = (_Float16)in[i];
}

// ---------------- pack W [K,N] row-major f32 into WMMA B fragments (f16) ----------------
// fragment f = kt*NT + nt holds the 32x16 tile; per lane: N = nt*16 + (lane&15),
// K = kt*32 + (lane>>4)*8 + (i&7) + (i>>3)*16, i = 0..15 (matches A table order).
__global__ void pack_b_kernel(const float* __restrict__ W, _Float16* __restrict__ Bp, int K, int N) {
  int idx = blockIdx.x * blockDim.x + threadIdx.x;
  if (idx >= K * N) return;
  int i    = idx & 15;
  int lane = (idx >> 4) & 31;
  int f    = idx >> 9;
  int NT   = N >> 4;
  int kt = f / NT, nt = f - kt * NT;
  int n = nt * 16 + (lane & 15);
  int k = kt * 32 + (lane >> 4) * 8 + (i & 7) + (i >> 3) * 16;
  Bp[idx] = (_Float16)W[(size_t)k * N + n];
}

// ---------------- WMMA GEMM: C[M, NT*16] = A[M,K](f16 row-major) * Bpacked ----------------
// One wave computes a 16 x (NT*16) strip: load A fragment once per K-step,
// issue NT back-to-back WMMAs against the NT column fragments (4x less A traffic).
template <int NT>
__global__ void wmma_gemm_kernel(const _Float16* __restrict__ A, const _Float16* __restrict__ Bp,
                                 float* __restrict__ C, int Mm, int Kk) {
  constexpr int Nn = NT * 16;
  int lane = threadIdx.x & 31;
  int wave = threadIdx.x >> 5;
  int mt = blockIdx.x * (blockDim.x >> 5) + wave;
  if (mt >= (Mm >> 4)) return;

  // A fragment addressing (ISA 16-bit A 16x32 layout):
  // lanes 0-15: M=lane,    halves 0..7 -> K=0..7,  halves 8..15 -> K=16..23
  // lanes16-31: M=lane-16, halves 0..7 -> K=8..15, halves 8..15 -> K=24..31
  int row = mt * 16 + (lane & 15);
  int kb  = (lane >> 4) * 8;
  const _Float16* arow  = A + (size_t)row * Kk + kb;
  const _Float16* bbase = Bp + (size_t)lane * 16;

  v8f acc[NT] = {};
  int KT = Kk >> 5;
  for (int kt = 0; kt < KT; ++kt) {
    __builtin_prefetch(arow + kt * 32 + 512, 0, 1);  // stream-ahead on A (global_prefetch_b8)
    v8h lo = *(const v8h*)(arow + kt * 32);
    v8h hi = *(const v8h*)(arow + kt * 32 + 16);
    v16h a;
#pragma unroll
    for (int i = 0; i < 8; ++i) { a[i] = lo[i]; a[i + 8] = hi[i]; }
    const _Float16* bk = bbase + (((size_t)kt * NT) << 9);
#pragma unroll
    for (int nt = 0; nt < NT; ++nt) {
      v16h b = *(const v16h*)(bk + ((size_t)nt << 9));
      acc[nt] = __builtin_amdgcn_wmma_f32_16x16x32_f16(false, a, false, b, (short)0,
                                                       acc[nt], false, false);
    }
  }
  // C/D layout: VGPR r, lanes 0-15 -> (M=r, N=lane); lanes 16-31 -> (M=8+r, N=lane-16)
  int crow0 = mt * 16 + (lane >> 4) * 8;
  int ccol  = lane & 15;
#pragma unroll
  for (int nt = 0; nt < NT; ++nt)
#pragma unroll
    for (int r = 0; r < 8; ++r)
      C[(size_t)(crow0 + r) * Nn + nt * 16 + ccol] = acc[nt][r];
}

// ---------------- per-(node,head) attention coefficients ----------------
template <int HH, int C>
__global__ void att_coef_kernel(const float* __restrict__ Hm, const float* __restrict__ att_s,
                                const float* __restrict__ att_d, float* __restrict__ a_s,
                                float* __restrict__ a_d, int N) {
  int idx = blockIdx.x * blockDim.x + threadIdx.x;
  if (idx >= N * HH) return;
  int h = idx & (HH - 1);
  const float* hp = Hm + (size_t)idx * C;
  float s = 0.f, d = 0.f;
#pragma unroll
  for (int c = 0; c < C; ++c) {
    s = fmaf(hp[c], att_s[h * C + c], s);
    d = fmaf(hp[c], att_d[h * C + c], d);
  }
  a_s[idx] = s;
  a_d[idx] = d;
}

template <int HH>
static __device__ __forceinline__ float edge_e(const int* src, const int* dst,
                                               const float* a_s, const float* a_d,
                                               long e, int h, int& sN, int& dN) {
  sN = src[e]; dN = dst[e];
  float v = a_s[(size_t)sN * HH + h] + a_d[(size_t)dN * HH + h];
  return v > 0.f ? v : NEG_SLOPE * v;
}

// ---------------- edge pass 1: segment max ----------------
template <int HH>
__global__ void edge_max_kernel(const int* __restrict__ src, const int* __restrict__ dst,
                                const float* __restrict__ a_s, const float* __restrict__ a_d,
                                unsigned* __restrict__ emax, long E) {
  long idx = (long)blockIdx.x * blockDim.x + threadIdx.x;
  if (idx >= E * HH) return;
  long e = idx >> __builtin_ctz(HH);
  int h = (int)(idx & (HH - 1));
  int s, d;
  float v = edge_e<HH>(src, dst, a_s, a_d, e, h, s, d);
  atomicMax(&emax[(size_t)d * HH + h], f2o(v));
}

// ---------------- edge pass 2: softmax denominator ----------------
template <int HH>
__global__ void edge_denom_kernel(const int* __restrict__ src, const int* __restrict__ dst,
                                  const float* __restrict__ a_s, const float* __restrict__ a_d,
                                  const unsigned* __restrict__ emax, float* __restrict__ denom,
                                  long E) {
  long idx = (long)blockIdx.x * blockDim.x + threadIdx.x;
  if (idx >= E * HH) return;
  long e = idx >> __builtin_ctz(HH);
  int h = (int)(idx & (HH - 1));
  int s, d;
  float v = edge_e<HH>(src, dst, a_s, a_d, e, h, s, d);
  float m = o2f(emax[(size_t)d * HH + h]);
  atomicAdd(&denom[(size_t)d * HH + h], __expf(v - m));
}

// ---------------- denom -> reciprocal, in place ----------------
__global__ void rcp_kernel(float* __restrict__ denom, long n) {
  long i = (long)blockIdx.x * blockDim.x + threadIdx.x;
  if (i < n) denom[i] = 1.f / (denom[i] + 1e-16f);
}

// ---------------- edge pass 3: alpha-weighted message scatter ----------------
// thread per (edge, head, channel-chunk of 8); CC = C/8 chunks per head
template <int HH, int C>
__global__ void edge_scatter_kernel(const int* __restrict__ src, const int* __restrict__ dst,
                                    const float* __restrict__ a_s, const float* __restrict__ a_d,
                                    const unsigned* __restrict__ emax, const float* __restrict__ rdenom,
                                    const float* __restrict__ Hm, float* __restrict__ out,
                                    long E) {
  constexpr int CC = C / 8;           // chunks of 8 channels
  constexpr int TPE = HH * CC;        // threads per edge (power of 2 here)
  long idx = (long)blockIdx.x * blockDim.x + threadIdx.x;
  if (idx >= E * TPE) return;
  long e = idx >> __builtin_ctz(TPE);
  int sub = (int)(idx & (TPE - 1));
  int h  = sub / CC;
  int cc = sub - h * CC;
  int s, d;
  float v = edge_e<HH>(src, dst, a_s, a_d, e, h, s, d);
  float m = o2f(emax[(size_t)d * HH + h]);
  float alpha = __expf(v - m) * rdenom[(size_t)d * HH + h];
  const float4* hp4 = (const float4*)(Hm + ((size_t)s * HH + h) * C + cc * 8);
  float* op = out + ((size_t)d * HH + h) * C + cc * 8;
  float4 m0 = hp4[0];
  float4 m1 = hp4[1];
  atomicAdd(&op[0], m0.x * alpha);
  atomicAdd(&op[1], m0.y * alpha);
  atomicAdd(&op[2], m0.z * alpha);
  atomicAdd(&op[3], m0.w * alpha);
  atomicAdd(&op[4], m1.x * alpha);
  atomicAdd(&op[5], m1.y * alpha);
  atomicAdd(&op[6], m1.z * alpha);
  atomicAdd(&op[7], m1.w * alpha);
}

// ---------------- bias + ELU + convert to f16 for next GEMM ----------------
template <int D>
__global__ void bias_elu_cvt_kernel(const float* __restrict__ aggr, const float* __restrict__ bias,
                                    _Float16* __restrict__ outH, long n) {
  long i = (long)blockIdx.x * blockDim.x + threadIdx.x;
  if (i >= n) return;
  float v = aggr[i] + bias[(int)(i & (D - 1))];
  v = v > 0.f ? v : (__expf(v) - 1.f);
  outH[i] = (_Float16)v;
}

// ---------------- final: bias + log_softmax over D channels, in place ----------------
template <int D>
__global__ void logsoftmax_bias_kernel(float* __restrict__ out, const float* __restrict__ bias,
                                       int N) {
  int n = blockIdx.x * blockDim.x + threadIdx.x;
  if (n >= N) return;
  float* p = out + (size_t)n * D;
  float mx = -3.402823466e38f;
  for (int c = 0; c < D; ++c) mx = fmaxf(mx, p[c] + bias[c]);
  float s = 0.f;
  for (int c = 0; c < D; ++c) s += __expf(p[c] + bias[c] - mx);
  float l = mx + __logf(s);
  for (int c = 0; c < D; ++c) p[c] = p[c] + bias[c] - l;
}

extern "C" void kernel_launch(void* const* d_in, const int* in_sizes, int n_in,
                              void* d_out, int out_size, void* d_ws, size_t ws_size,
                              hipStream_t stream) {
  const float* x      = (const float*)d_in[0];
  const int*   ei     = (const int*)d_in[1];
  const float* W1     = (const float*)d_in[2];
  const float* att_s1 = (const float*)d_in[3];
  const float* att_d1 = (const float*)d_in[4];
  const float* bias1  = (const float*)d_in[5];
  const float* W2     = (const float*)d_in[6];
  const float* att_s2 = (const float*)d_in[7];
  const float* att_d2 = (const float*)d_in[8];
  const float* bias2  = (const float*)d_in[9];

  const long N = in_sizes[0] / F_IN;
  const long E = in_sizes[1] / 2;
  const int* src = ei;
  const int* dst = ei + E;

  char* w = (char*)d_ws;
  auto carve = [&](size_t bytes) -> char* {
    char* p = w;
    w += (bytes + 255) & ~(size_t)255;
    return p;
  };
  _Float16* xh  = (_Float16*)carve((size_t)N * F_IN * 2);
  _Float16* W1p = (_Float16*)carve((size_t)F_IN * D1 * 2);
  float*    h1  = (float*)carve((size_t)N * D1 * 4);
  float*    as1 = (float*)carve((size_t)N * HEADS1 * 4);
  float*    ad1 = (float*)carve((size_t)N * HEADS1 * 4);
  unsigned* em1 = (unsigned*)carve((size_t)N * HEADS1 * 4);
  float*    dn1 = (float*)carve((size_t)N * HEADS1 * 4);
  float*    ag1 = (float*)carve((size_t)N * D1 * 4);
  _Float16* a1h = (_Float16*)carve((size_t)N * D1 * 2);
  _Float16* W2p = (_Float16*)carve((size_t)D1 * D2 * 2);
  float*    h2  = (float*)carve((size_t)N * D2 * 4);
  float*    as2 = (float*)carve((size_t)N * 4);
  float*    ad2 = (float*)carve((size_t)N * 4);
  unsigned* em2 = (unsigned*)carve((size_t)N * 4);
  float*    dn2 = (float*)carve((size_t)N * 4);
  float*    out = (float*)d_out;   // layer-2 aggregation buffer, finalized in place

  hipMemsetAsync(em1, 0, (size_t)N * HEADS1 * 4, stream);  // 0 is below every f2o() value
  hipMemsetAsync(dn1, 0, (size_t)N * HEADS1 * 4, stream);
  hipMemsetAsync(ag1, 0, (size_t)N * D1 * 4, stream);
  hipMemsetAsync(em2, 0, (size_t)N * 4, stream);
  hipMemsetAsync(dn2, 0, (size_t)N * 4, stream);
  hipMemsetAsync(out, 0, (size_t)N * D2 * 4, stream);

  const int TB = 256;

  long nx = N * F_IN;
  cvt_f16_kernel<<<(unsigned)((nx + TB - 1) / TB), TB, 0, stream>>>(x, xh, nx);
  pack_b_kernel<<<(F_IN * D1 + TB - 1) / TB, TB, 0, stream>>>(W1, W1p, F_IN, D1);
  pack_b_kernel<<<(D1 * D2 + TB - 1) / TB, TB, 0, stream>>>(W2, W2p, D1, D2);

  // ----- layer 1 -----
  long mtiles = N / 16;  // one 16x64 strip per wave, 8 waves per block
  wmma_gemm_kernel<D1 / 16><<<(unsigned)((mtiles + 7) / 8), TB, 0, stream>>>(xh, W1p, h1,
                                                                             (int)N, F_IN);

  att_coef_kernel<HEADS1, CH1><<<(unsigned)((N * HEADS1 + TB - 1) / TB), TB, 0, stream>>>(
      h1, att_s1, att_d1, as1, ad1, (int)N);

  long eh1 = E * HEADS1;
  unsigned gE1 = (unsigned)((eh1 + TB - 1) / TB);
  edge_max_kernel<HEADS1><<<gE1, TB, 0, stream>>>(src, dst, as1, ad1, em1, E);
  edge_denom_kernel<HEADS1><<<gE1, TB, 0, stream>>>(src, dst, as1, ad1, em1, dn1, E);
  rcp_kernel<<<(unsigned)((N * HEADS1 + TB - 1) / TB), TB, 0, stream>>>(dn1, N * HEADS1);
  edge_scatter_kernel<HEADS1, CH1><<<gE1, TB, 0, stream>>>(src, dst, as1, ad1, em1, dn1, h1, ag1, E);

  long nd1 = N * D1;
  bias_elu_cvt_kernel<D1><<<(unsigned)((nd1 + TB - 1) / TB), TB, 0, stream>>>(ag1, bias1, a1h, nd1);

  // ----- layer 2 -----
  wmma_gemm_kernel<D2 / 16><<<(unsigned)((mtiles + 7) / 8), TB, 0, stream>>>(a1h, W2p, h2,
                                                                             (int)N, D1);

  att_coef_kernel<1, D2><<<(unsigned)((N + TB - 1) / TB), TB, 0, stream>>>(
      h2, att_s2, att_d2, as2, ad2, (int)N);

  unsigned gE2 = (unsigned)((E + TB - 1) / TB);
  edge_max_kernel<1><<<gE2, TB, 0, stream>>>(src, dst, as2, ad2, em2, E);
  edge_denom_kernel<1><<<gE2, TB, 0, stream>>>(src, dst, as2, ad2, em2, dn2, E);
  rcp_kernel<<<(unsigned)((N + TB - 1) / TB), TB, 0, stream>>>(dn2, N);
  // layer 2 scatter: 8 chunk-threads per edge (HH=1, C=64 -> 8 chunks)
  unsigned gS2 = (unsigned)((E * 8 + TB - 1) / TB);
  edge_scatter_kernel<1, D2><<<gS2, TB, 0, stream>>>(src, dst, as2, ad2, em2, dn2, h2, out, E);

  logsoftmax_bias_kernel<D2><<<(unsigned)((N + TB - 1) / TB), TB, 0, stream>>>(out, bias2, (int)N);
}
